// MultiHeadSelfAttention_77704548319830
// MI455X (gfx1250) — matrix operand order repaired and checked by reading, not compiled, not run on hardware
//
#include <hip/hip_runtime.h>
#include <hip/hip_bf16.h>
#include <cstdint>

// MHA forward for B=2,S=2048,D=1024,H=8,Hd=128 on gfx1250 (wave32, WMMA).
// Stage 1: QKV projection GEMMs, f32 -> f16 via LDS, WMMA f16->f32 accum.
// Stage 2: fused flash attention, software-pipelined: K tile staged with
//          async global->LDS DMA (ASYNCcnt) double-buffered under compute.

typedef __attribute__((ext_vector_type(16))) _Float16 v16h;
typedef __attribute__((ext_vector_type(8)))  _Float16 h8;
typedef __attribute__((ext_vector_type(4)))  _Float16 h4;
typedef __attribute__((ext_vector_type(8)))  float    v8f;
typedef int v4i __attribute__((vector_size(16)));

#define B_  2
#define S_  2048
#define D_  1024
#define H_  8
#define HD_ 128

static __device__ __forceinline__ v8f wmma16(v16h a, v16h b, v8f c) {
  return __builtin_amdgcn_wmma_f32_16x16x32_f16(false, a, false, b,
                                                (short)0, c, false, false);
}

static __device__ __forceinline__ v16h cat8(h8 lo, h8 hi) {
  return __builtin_shufflevector(lo, hi, 0, 1, 2, 3, 4, 5, 6, 7,
                                 8, 9, 10, 11, 12, 13, 14, 15);
}

// ---- CDNA5 async global->LDS copy (ASYNCcnt-tracked), with asm fallback ----
static __device__ __forceinline__ void async_b128(const void* g, void* lds) {
#if __has_builtin(__builtin_amdgcn_global_load_async_to_lds_b128)
  __builtin_amdgcn_global_load_async_to_lds_b128(
      (__attribute__((address_space(1))) v4i*)(uintptr_t)g,
      (__attribute__((address_space(3))) v4i*)(unsigned)(uintptr_t)lds,
      0, 0);
#else
  asm volatile("global_load_async_to_lds_b128 %0, %1, off"
               :: "v"((unsigned)(uintptr_t)lds), "v"(g)
               : "memory");
#endif
}

static __device__ __forceinline__ void wait_async0() {
#if __has_builtin(__builtin_amdgcn_s_wait_asynccnt)
  __builtin_amdgcn_s_wait_asynccnt(0);
#else
  asm volatile("s_wait_asynccnt 0" ::: "memory");
#endif
}

static __device__ __forceinline__ void wait_async2() {
  // Async loads complete in order: <=2 outstanding means the current tile's
  // two copies are done while the next tile's two stay in flight.
#if __has_builtin(__builtin_amdgcn_s_wait_asynccnt)
  __builtin_amdgcn_s_wait_asynccnt(2);
#else
  asm volatile("s_wait_asynccnt 2" ::: "memory");
#endif
}

static __device__ __forceinline__ void lds_fence() {
  // P tile is wave-private; LDS ops from one wave execute in order, so a
  // compiler barrier + dscnt drain is sufficient (no block barrier needed).
  asm volatile("s_wait_dscnt 0" ::: "memory");
}

// ---------------------------------------------------------------------------
// Kernel 1: X @ W + bias for X in {q,k,v}; output f16 in [B,H,S,Hd] layout.
// Block = 256 threads (8 waves). Block tile = 32 (M) x 128 (N); each wave owns
// a 16x32 sub-tile (2 WMMAs per K-step). B tile stored TRANSPOSED in LDS so
// B-fragments are contiguous ds_load_b128 pairs.
// ---------------------------------------------------------------------------
__global__ __launch_bounds__(256) void qkv_proj_kernel(
    const float* __restrict__ q, const float* __restrict__ k,
    const float* __restrict__ v,
    const float* __restrict__ Wq, const float* __restrict__ bq,
    const float* __restrict__ Wk, const float* __restrict__ bk,
    const float* __restrict__ Wv, const float* __restrict__ bv,
    _Float16* __restrict__ Qh, _Float16* __restrict__ Kh,
    _Float16* __restrict__ Vh)
{
  const int z = blockIdx.z;
  const float* __restrict__ X    = (z == 0) ? q  : ((z == 1) ? k  : v);
  const float* __restrict__ W    = (z == 0) ? Wq : ((z == 1) ? Wk : Wv);
  const float* __restrict__ bias = (z == 0) ? bq : ((z == 1) ? bk : bv);
  _Float16* __restrict__ O       = (z == 0) ? Qh : ((z == 1) ? Kh : Vh);

  __shared__ __align__(16) _Float16 As[32][32];   // 32 M x 32 K (2 KB)
  __shared__ __align__(16) _Float16 Bt[128][40];  // transposed [N][K], padded (10 KB)

  const int t    = threadIdx.x;
  const int lane = t & 31;
  const int wave = t >> 5;         // 0..7
  const int g    = lane >> 4;      // half-wave group
  const int ln   = lane & 15;

  const int M0 = blockIdx.x * 32;
  const int N0 = blockIdx.y * 128;
  const int mi = wave & 1;         // M sub-tile (0/1)
  const int ni = wave >> 1;        // N sub-tile (0..3), 32 cols each

  v8f c0 = {}, c1 = {};

  for (int kb = 0; kb < D_; kb += 32) {
    __syncthreads();
    // --- stage A tile: 32x32 f32 -> f16, 4 elems/thread, one h4 (b64) store
    {
      const int idx = t * 4;
      const int m   = idx >> 5;
      const int kk  = idx & 31;
      const float4 a4 = *(const float4*)(X + (size_t)(M0 + m) * D_ + kb + kk);
      h4 hv = {(_Float16)a4.x, (_Float16)a4.y, (_Float16)a4.z, (_Float16)a4.w};
      *(h4*)&As[m][kk] = hv;
    }
    // --- stage B tile transposed: 32 K x 128 N, 16 elems/thread
    {
      const int kr = t >> 3;            // K row 0..31
      const int n0 = (t & 7) * 16;      // N col base
      const float* src = W + (size_t)(kb + kr) * D_ + N0 + n0;
#pragma unroll
      for (int j = 0; j < 16; j += 4) {
        const float4 w4 = *(const float4*)(src + j);
        Bt[n0 + j + 0][kr] = (_Float16)w4.x;
        Bt[n0 + j + 1][kr] = (_Float16)w4.y;
        Bt[n0 + j + 2][kr] = (_Float16)w4.z;
        Bt[n0 + j + 3][kr] = (_Float16)w4.w;
      }
    }
    __syncthreads();

    // --- A fragment: row M = mi*16+ln; K halves at g*8 and 16+g*8
    const v16h a = cat8(*(const h8*)&As[mi * 16 + ln][g * 8],
                        *(const h8*)&As[mi * 16 + ln][16 + g * 8]);
    // --- two B fragments (contiguous K in Bt rows)
    {
      const int n = ni * 32 + ln;
      const v16h bf = cat8(*(const h8*)&Bt[n][g * 8],
                           *(const h8*)&Bt[n][16 + g * 8]);
      c0 = wmma16(a, bf, c0);
    }
    {
      const int n = ni * 32 + 16 + ln;
      const v16h bf = cat8(*(const h8*)&Bt[n][g * 8],
                           *(const h8*)&Bt[n][16 + g * 8]);
      c1 = wmma16(a, bf, c1);
    }
  }

  // --- bias add + store f16 to [B,H,S,Hd] layout
#pragma unroll
  for (int sub = 0; sub < 2; ++sub) {
    const int n  = N0 + ni * 32 + sub * 16 + ln;
    const float bn = bias[n];
    const int h  = n >> 7;
    const int hd = n & 127;
    const v8f cc = sub ? c1 : c0;
#pragma unroll
    for (int vv = 0; vv < 8; ++vv) {
      const int m = M0 + mi * 16 + vv + 8 * g;
      const int b = m >> 11;
      const int s = m & 2047;
      O[(((size_t)(b * H_ + h) * S_ + s) * HD_ + hd)] = (_Float16)(cc[vv] + bn);
    }
  }
}

// ---------------------------------------------------------------------------
// Kernel 2: fused flash attention, software-pipelined with double-buffered
// LDS. Block = 256 threads (8 waves) = 128 query rows of one (b,h).
// Tile i+1's K DMA + V register prefetch overlap tile i's 16 WMMAs + softmax.
// ---------------------------------------------------------------------------
__global__ __launch_bounds__(256) void flash_attn_kernel(
    const _Float16* __restrict__ Qh, const _Float16* __restrict__ Kh,
    const _Float16* __restrict__ Vh, const unsigned char* __restrict__ mask,
    float* __restrict__ out)
{
  const int bh = blockIdx.y;           // 0..15 -> (b,h)
  const int b  = bh >> 3;
  const int h  = bh & 7;
  const int q0 = blockIdx.x * 128;

  const int t    = threadIdx.x;
  const int lane = t & 31;
  const int wave = t >> 5;
  const int g    = lane >> 4;
  const int ln   = lane & 15;

  __shared__ __align__(16) _Float16 Ks[2][32][136];   // K tiles [key][hd] (17 KB)
  __shared__ __align__(16) _Float16 Vt[2][HD_][40];   // V tiles transposed (20 KB)
  __shared__ __align__(16) _Float16 Ps[8][16][32];    // per-wave P tiles (8 KB)

  const _Float16* __restrict__ Qb = Qh + (size_t)bh * S_ * HD_;
  const _Float16* __restrict__ Kb = Kh + (size_t)bh * S_ * HD_;
  const _Float16* __restrict__ Vb = Vh + (size_t)bh * S_ * HD_;
  const unsigned char* __restrict__ mb = mask + (size_t)b * S_ * S_;

  const int qr = q0 + wave * 16;

  // --- Q A-fragments, pre-scaled by 1/sqrt(128) (folds the score scaling)
  const _Float16 hsc = (_Float16)0.08838834764831845f;
  v16h qa[4];
#pragma unroll
  for (int ch = 0; ch < 4; ++ch) {
    const _Float16* src = Qb + (size_t)(qr + ln) * HD_ + ch * 32 + g * 8;
    qa[ch] = cat8(*(const h8*)src, *(const h8*)(src + 16)) * hsc;
  }
  const float MASKED = -88388.34764831845f;   // -1e6 / sqrt(128)

  v8f acc[8];
#pragma unroll
  for (int nn = 0; nn < 8; ++nn) acc[nn] = v8f{};
  float mrow[8], lrow[8];
#pragma unroll
  for (int vv = 0; vv < 8; ++vv) { mrow[vv] = -1e30f; lrow[vv] = 0.0f; }

  const int key = t >> 3;                 // staging: key row 0..31
  const int hd0 = (t & 7) * 16;           // staging: hd base

  // --- prologue: start tile 0's K DMA, load tile 0's V rows into registers
  h8 va0, va1;
  {
    const _Float16* ksrc = Kb + (size_t)key * HD_ + hd0;
    async_b128(ksrc,     &Ks[0][key][hd0]);
    async_b128(ksrc + 8, &Ks[0][key][hd0 + 8]);
    const _Float16* vsrc = Vb + (size_t)key * HD_ + hd0;
    va0 = *(const h8*)vsrc;
    va1 = *(const h8*)(vsrc + 8);
  }

  for (int it = 0; it < S_ / 32; ++it) {
    const int k0   = it * 32;
    const int cur  = it & 1;
    const int nxt  = cur ^ 1;
    const bool last = (it == S_ / 32 - 1);

    __syncthreads();   // iter it-1's readers of Ks[nxt]/Vt[cur] are done

    // --- issue next tile's K DMA into the other buffer (overlaps compute)
    if (!last) {
      const _Float16* ksrc = Kb + (size_t)(k0 + 32 + key) * HD_ + hd0;
      async_b128(ksrc,     &Ks[nxt][key][hd0]);
      async_b128(ksrc + 8, &Ks[nxt][key][hd0 + 8]);
    }
    // --- transpose-store this tile's V registers into Vt[cur]
#pragma unroll
    for (int i = 0; i < 8; ++i) {
      Vt[cur][hd0 + i][key]     = va0[i];
      Vt[cur][hd0 + 8 + i][key] = va1[i];
    }
    // --- prefetch next tile's V rows into registers
    if (!last) {
      const _Float16* vsrc = Vb + (size_t)(k0 + 32 + key) * HD_ + hd0;
      va0 = *(const h8*)vsrc;
      va1 = *(const h8*)(vsrc + 8);
    }

    if (!last) wait_async2(); else wait_async0();   // current K tile landed
    __syncthreads();                                // Ks[cur] + Vt[cur] visible

    // --- scores: two 16-key sub-tiles from Ks[cur]
    v8f cL = {}, cR = {};
#pragma unroll
    for (int ch = 0; ch < 4; ++ch) {
      const int co = ch * 32 + g * 8;
      const v16h kfL = cat8(*(const h8*)&Ks[cur][ln][co],
                            *(const h8*)&Ks[cur][ln][co + 16]);
      const v16h kfR = cat8(*(const h8*)&Ks[cur][16 + ln][co],
                            *(const h8*)&Ks[cur][16 + ln][co + 16]);
      cL = wmma16(qa[ch], kfL, cL);
      cR = wmma16(qa[ch], kfR, cR);
    }

    // --- mask (reference: where(mask,-1e6) before scaling; scaling folded
    //     into Q, so masked lanes get -1e6/sqrt(128)) + online softmax
    float alpha[8], pL[8], pR[8];
#pragma unroll
    for (int vv = 0; vv < 8; ++vv) {
      const int qrow = qr + vv + 8 * g;
      const unsigned char* mp = mb + (size_t)qrow * S_ + k0;
      const float sL = mp[ln]      ? MASKED : cL[vv];
      const float sR = mp[16 + ln] ? MASKED : cR[vv];
      float tmax = fmaxf(sL, sR);
#pragma unroll
      for (int d = 1; d < 16; d <<= 1) tmax = fmaxf(tmax, __shfl_xor(tmax, d, 32));
      const float mnew = fmaxf(mrow[vv], tmax);
      const float a_   = __expf(mrow[vv] - mnew);
      const float eL   = __expf(sL - mnew);
      const float eR   = __expf(sR - mnew);
      float rs = eL + eR;
#pragma unroll
      for (int d = 1; d < 16; d <<= 1) rs += __shfl_xor(rs, d, 32);
      lrow[vv] = lrow[vv] * a_ + rs;
      mrow[vv] = mnew;
      alpha[vv] = a_;
      pL[vv] = eL; pR[vv] = eR;
    }
#pragma unroll
    for (int nn = 0; nn < 8; ++nn)
#pragma unroll
      for (int vv = 0; vv < 8; ++vv) acc[nn][vv] *= alpha[vv];

    // --- P (C-frag layout) -> wave-private LDS -> A-frag layout
#pragma unroll
    for (int vv = 0; vv < 8; ++vv) {
      Ps[wave][vv + 8 * g][ln]      = (_Float16)pL[vv];
      Ps[wave][vv + 8 * g][16 + ln] = (_Float16)pR[vv];
    }
    lds_fence();   // same-wave LDS is in-order; just stop compiler reordering

    const v16h pa = cat8(*(const h8*)&Ps[wave][ln][g * 8],
                         *(const h8*)&Ps[wave][ln][16 + g * 8]);

    // --- PV: acc[n] += P(16x32) * V(32x16) for 8 hd tiles
#pragma unroll
    for (int nn = 0; nn < 8; ++nn) {
      const int hd = nn * 16 + ln;
      const v16h vf = cat8(*(const h8*)&Vt[cur][hd][g * 8],
                           *(const h8*)&Vt[cur][hd][16 + g * 8]);
      acc[nn] = wmma16(pa, vf, acc[nn]);
    }
  }

  // --- normalize (rcp + mul instead of 64 divides) and store f32 [B,S,D]
  float rinv[8];
#pragma unroll
  for (int vv = 0; vv < 8; ++vv) rinv[vv] = __builtin_amdgcn_rcpf(lrow[vv]);

  float* __restrict__ ob = out + (size_t)b * S_ * D_;
#pragma unroll
  for (int nn = 0; nn < 8; ++nn) {
    const int hd = h * HD_ + nn * 16 + ln;
#pragma unroll
    for (int vv = 0; vv < 8; ++vv) {
      const int s = qr + vv + 8 * g;
      ob[(size_t)s * D_ + hd] = acc[nn][vv] * rinv[vv];
    }
  }
}

// ---------------------------------------------------------------------------
extern "C" void kernel_launch(void* const* d_in, const int* in_sizes, int n_in,
                              void* d_out, int out_size, void* d_ws, size_t ws_size,
                              hipStream_t stream) {
  (void)in_sizes; (void)n_in; (void)out_size; (void)ws_size;

  const float* q  = (const float*)d_in[0];
  const float* k  = (const float*)d_in[1];
  const float* v  = (const float*)d_in[2];
  const unsigned char* mask = (const unsigned char*)d_in[3];  // jax bool -> 1 byte
  const float* Wq = (const float*)d_in[4];
  const float* bq = (const float*)d_in[5];
  const float* Wk = (const float*)d_in[6];
  const float* bk = (const float*)d_in[7];
  const float* Wv = (const float*)d_in[8];
  const float* bv = (const float*)d_in[9];
  float* out = (float*)d_out;

  // Workspace: Q/K/V projected to f16 in [B,H,S,Hd] layout, 8 MB each.
  _Float16* Qh = (_Float16*)d_ws;
  _Float16* Kh = Qh + (size_t)B_ * S_ * D_;
  _Float16* Vh = Kh + (size_t)B_ * S_ * D_;

  qkv_proj_kernel<<<dim3((B_ * S_) / 32, D_ / 128, 3), dim3(256), 0, stream>>>(
      q, k, v, Wq, bq, Wk, bk, Wv, bv, Qh, Kh, Vh);

  flash_attn_kernel<<<dim3(S_ / 128, B_ * H_), dim3(256), 0, stream>>>(
      Qh, Kh, Vh, mask, out);
}
